// Decoder_2800318677715
// MI455X (gfx1250) — matrix operand order repaired and checked by reading, not compiled, not run on hardware
//
#include <hip/hip_runtime.h>

// ---------------- dims ----------------
constexpr int kT = 64, kB = 64, kS = 100;
constexpr int kDw = 512, kE = 512, kH = 512, kA = 512;
constexpr int kH3 = 3 * kH;            // 1536

typedef __attribute__((ext_vector_type(16))) __bf16 v16bf;
typedef __attribute__((ext_vector_type(8)))  __bf16 v8bf;
typedef __attribute__((ext_vector_type(8)))  float  v8f;

__device__ __forceinline__ int imin(int a, int b) { return a < b ? a : b; }
__device__ __forceinline__ int imax(int a, int b) { return a > b ? a : b; }

__device__ __forceinline__ v8f zero8() {
  v8f z;
#pragma unroll
  for (int i = 0; i < 8; ++i) z[i] = 0.0f;
  return z;
}

__device__ __forceinline__ v8f wmma_bf16(v16bf a, v16bf b, v8f c) {
  // D = A(16x32 bf16) * B(32x16 bf16) + C(16x16 f32)
  return __builtin_amdgcn_wmma_f32_16x16x32_bf16(false, a, false, b, (short)0, c,
                                                 false, false);
}

__device__ __forceinline__ float sigm(float x) { return 1.0f / (1.0f + __expf(-x)); }
__device__ __forceinline__ float tanh_f(float x) {
  return 1.0f - 2.0f / (__expf(2.0f * x) + 1.0f);
}

// A-fragment, 16x32 bf16, ISA layout: lane l = row (l&15);
// halves 0..7  -> K = 8*(l>>4) + j ; halves 8..15 -> K = 16 + 8*(l>>4) + (j-8)
// Abase points at (row 0, k); rows are 512-element slices.
__device__ __forceinline__ v16bf load_afrag(const __bf16* Abase, int row, int lane) {
  const __bf16* p = Abase + ((size_t)row << 9) + ((lane >> 4) << 3);
  v8bf lo = *(const v8bf*)p;
  v8bf hi = *(const v8bf*)(p + 16);
  v16bf a;
#pragma unroll
  for (int i = 0; i < 8; ++i) { a[i] = lo[i]; a[i + 8] = hi[i]; }
  return a;
}

// B-fragment, 32x16 bf16 from weight W stored (N,K) row-major (out = x @ W^T).
// Lane l holds column n0 + (l&15); K values k0 + 8*(l>>4) .. +8 and k0 + 16 + 8*(l>>4) .. +8.
__device__ __forceinline__ v16bf load_bfrag(const __bf16* W, int Kstride, int n0,
                                            int k0, int lane) {
  const __bf16* p = W + (size_t)(n0 + (lane & 15)) * Kstride + k0 + ((lane >> 4) << 3);
  v8bf b0 = *(const v8bf*)p;
  v8bf b1 = *(const v8bf*)(p + 16);
  v16bf b;
#pragma unroll
  for (int i = 0; i < 8; ++i) { b[i] = b0[i]; b[i + 8] = b1[i]; }
  return b;
}

// One contiguous K-segment of a 16x64 output tile (4 accumulators).
// Aseg points at (row 0, segment start); kofs is the segment's offset in W's K.
__device__ __forceinline__ void mm4(const __bf16* Aseg, int klen,
                                    const __bf16* __restrict__ W, int Kstride, int kofs,
                                    int row, int lane, int n0,
                                    v8f& a0, v8f& a1, v8f& a2, v8f& a3) {
#pragma unroll 2
  for (int kk = 0; kk < klen; kk += 32) {
    v16bf a  = load_afrag(Aseg + kk, row, lane);
    int k = kofs + kk;
    v16bf b0 = load_bfrag(W, Kstride, n0,      k, lane);
    v16bf b1 = load_bfrag(W, Kstride, n0 + 16, k, lane);
    v16bf b2 = load_bfrag(W, Kstride, n0 + 32, k, lane);
    v16bf b3 = load_bfrag(W, Kstride, n0 + 48, k, lane);
    a0 = wmma_bf16(a, b0, a0);
    a1 = wmma_bf16(a, b1, a1);
    a2 = wmma_bf16(a, b2, a2);
    a3 = wmma_bf16(a, b3, a3);
  }
}

// One contiguous K-segment of a GRU tile: 3 gate column-groups x 4 subtiles,
// sharing one A fragment per K-chunk (12 wmma per chunk).
__device__ __forceinline__ void gru3x4(const __bf16* Aseg, int klen,
                                       const __bf16* __restrict__ W, int Kstride, int kofs,
                                       int row, int lane, int n0,
                                       v8f* g0, v8f* g1, v8f* g2) {
  for (int kk = 0; kk < klen; kk += 32) {
    v16bf a = load_afrag(Aseg + kk, row, lane);
    int k = kofs + kk;
#pragma unroll
    for (int g = 0; g < 3; ++g) {
      int nb = g * kH + n0;
      v16bf b0 = load_bfrag(W, Kstride, nb,      k, lane);
      v16bf b1 = load_bfrag(W, Kstride, nb + 16, k, lane);
      v16bf b2 = load_bfrag(W, Kstride, nb + 32, k, lane);
      v16bf b3 = load_bfrag(W, Kstride, nb + 48, k, lane);
      v8f* acc = (g == 0) ? g0 : (g == 1) ? g1 : g2;
      acc[0] = wmma_bf16(a, b0, acc[0]);
      acc[1] = wmma_bf16(a, b1, acc[1]);
      acc[2] = wmma_bf16(a, b2, acc[2]);
      acc[3] = wmma_bf16(a, b3, acc[3]);
    }
  }
}

// ---------------- prologue kernels ----------------
__global__ void k_conv_bf(const float* __restrict__ s, __bf16* __restrict__ d, size_t n) {
  size_t i = (size_t)blockIdx.x * 256 + threadIdx.x;
  if (i < n) d[i] = (__bf16)s[i];
}

__global__ void k_embed(const int* __restrict__ inp, const float* __restrict__ tab,
                        __bf16* __restrict__ emb) {
  size_t i = (size_t)blockIdx.x * 256 + threadIdx.x;
  if (i >= (size_t)kT * kB * kDw) return;
  size_t tb = i >> 9;
  int d = (int)(i & 511);
  int tok = inp[tb];
  emb[i] = (__bf16)tab[(size_t)tok * kDw + d];
}

__global__ void k_ctx_bf(const float* __restrict__ ctx, __bf16* __restrict__ dst) {
  size_t i = (size_t)blockIdx.x * 256 + threadIdx.x;
  if (i >= (size_t)kB * kS * kE) return;
  int b = (int)(i / (kS * kE));
  int rem = (int)(i % (kS * kE));
  int s = rem / kE, e = rem % kE;
  dst[i] = (__bf16)ctx[((size_t)s * kB + b) * kE + e];   // (S,B,E) -> (B,S,E)
}

__global__ void k_init(const float* __restrict__ hidden, const float* __restrict__ init_att,
                       float* h0f, float* h1f, __bf16* h0b, __bf16* h1b,
                       float* ctxf, __bf16* ctxb) {
  int i = blockIdx.x * 256 + threadIdx.x;
  if (i >= kB * kH) return;
  float a0 = hidden[i], a1 = hidden[kB * kH + i], c = init_att[i];
  h0f[i] = a0; h1f[i] = a1;
  h0b[i] = (__bf16)a0; h1b[i] = (__bf16)a1;
  ctxf[i] = c; ctxb[i] = (__bf16)c;
}

// ---------------- generic WMMA GEMM:  C[M,N] = [A0|A1|A2][M,K] @ W^T + bias --
// Wave computes a 16x64 tile; grid supplies (M/16)*(N/64) waves, 8 per block.
// K splits at k1 / k2 into up to three 512-wide row-slice sources.
__global__ void k_gemm(const __bf16* __restrict__ A0, const __bf16* __restrict__ A1,
                       const __bf16* __restrict__ A2, int k1, int k2, int K,
                       const __bf16* __restrict__ W, int N,
                       const float* __restrict__ bias, float* __restrict__ C) {
  int wid = (blockIdx.x * blockDim.x + threadIdx.x) >> 5;
  int lane = threadIdx.x & 31;
  int tn = N >> 6;
  int mi = wid / tn, ni = wid % tn;
  int row = (mi << 4) + (lane & 15);
  int n0 = ni << 6;
  v8f a0 = zero8(), a1 = zero8(), a2 = zero8(), a3 = zero8();
  int l0 = imin(k1, K);
  int l1 = imax(0, imin(k2, K) - l0);
  int l2 = K - l0 - l1;
  mm4(A0, l0, W, K, 0,       row, lane, n0, a0, a1, a2, a3);
  mm4(A1, l1, W, K, l0,      row, lane, n0, a0, a1, a2, a3);
  mm4(A2, l2, W, K, l0 + l1, row, lane, n0, a0, a1, a2, a3);
  int colb = n0 + (lane & 15);
  int rb = (mi << 4) + ((lane >> 4) << 3);
  v8f acc[4] = {a0, a1, a2, a3};
#pragma unroll
  for (int s = 0; s < 4; ++s) {
    int col = colb + (s << 4);
    float bv = bias ? bias[col] : 0.0f;
#pragma unroll
    for (int j = 0; j < 8; ++j)
      C[(size_t)(rb + j) * N + col] = acc[s][j] + bv;
  }
}

// ---------------- fused GRU layer ----------------
// gi = x @ W_ih^T ; gh = h @ W_hh^T ; gate math; writes h (f32 + bf16) and the
// g_hid / hid_f output slots.  32 waves (4 blocks x 256), each owns 16 rows x
// 64 H-columns -> 24 wmma per K-chunk across the two passes.
__global__ void k_gru(const __bf16* __restrict__ A0, const __bf16* __restrict__ A1,
                      int k1, int Kin,
                      const __bf16* __restrict__ Wih, const __bf16* __restrict__ Whh,
                      const float* __restrict__ b_ih, const float* __restrict__ b_hh,
                      const float* __restrict__ hprev, const __bf16* __restrict__ hprevbf,
                      float* __restrict__ hout, __bf16* __restrict__ houtbf,
                      float* __restrict__ ghid, float* __restrict__ hidf) {
  int wid = (blockIdx.x * blockDim.x + threadIdx.x) >> 5;   // 0..31
  int lane = threadIdx.x & 31;
  int mi = wid >> 3, nj = wid & 7;
  int row = (mi << 4) + (lane & 15);
  int n0 = nj << 6;
  v8f air[4], aiz[4], ain[4], ahn[4];
#pragma unroll
  for (int s = 0; s < 4; ++s) { air[s] = zero8(); aiz[s] = zero8(); ain[s] = zero8(); ahn[s] = zero8(); }

  // pass 1: gi (input path, K = Kin); A = [A0 | A1] split at k1
  int l0 = imin(k1, Kin);
  int l1 = Kin - l0;
  gru3x4(A0, l0, Wih, Kin, 0,  row, lane, n0, air, aiz, ain);
  gru3x4(A1, l1, Wih, Kin, l0, row, lane, n0, air, aiz, ain);
  // pass 2: gh (hidden path, K = 512); r,z fold into air/aiz, n kept separate
  gru3x4(hprevbf, kH, Whh, kH, 0, row, lane, n0, air, aiz, ahn);

  int colb = n0 + (lane & 15);
  int rb = (mi << 4) + ((lane >> 4) << 3);
#pragma unroll
  for (int s = 0; s < 4; ++s) {
    int col = colb + (s << 4);
#pragma unroll
    for (int j = 0; j < 8; ++j) {
      float ir = air[s][j] + b_ih[col] + b_hh[col];
      float iz = aiz[s][j] + b_ih[kH + col] + b_hh[kH + col];
      float rg = sigm(ir);
      float zg = sigm(iz);
      float nn = ain[s][j] + b_ih[2 * kH + col] + rg * (ahn[s][j] + b_hh[2 * kH + col]);
      float nt = tanh_f(nn);
      size_t off = (size_t)(rb + j) * kH + col;
      float hp = hprev[off];
      float h = (1.0f - zg) * nt + zg * hp;
      hout[off] = h;
      houtbf[off] = (__bf16)h;
      ghid[off] = h;
      hidf[off] = h;
    }
  }
}

// ---------------- fused attention / softmax / context / copy-gate ------------
__global__ void k_attn(int t, const float* __restrict__ pre, const float* __restrict__ qf,
                       const float* __restrict__ v_att, const float* __restrict__ mask,
                       const float* __restrict__ context, const float* __restrict__ h1f,
                       const float* __restrict__ Wcopy, const float* __restrict__ bcopy,
                       float* __restrict__ ctxf, __bf16* __restrict__ ctxbf,
                       float* __restrict__ o_cfull, float* __restrict__ o_clast,
                       float* __restrict__ o_ctxf, float* __restrict__ o_copy) {
  __shared__ float sq[kA];
  __shared__ float se[kS + 4];
  __shared__ float sr[128];
  int b = blockIdx.x, tid = threadIdx.x;
  for (int i = tid; i < kA; i += 128) sq[i] = qf[b * kA + i];
  __syncthreads();
  for (int s = tid; s < kS; s += 128) {
    const float* pp = pre + ((size_t)b * kS + s) * kA;
    float acc = 0.0f;
    for (int a = 0; a < kA; ++a)
      acc += v_att[a] * tanh_f(pp[a] + sq[a]);
    if (mask[b * kS + s] > 0.5f) acc = -1e18f;
    se[s] = acc;
  }
  __syncthreads();
  if (tid == 0) {
    float m = -3.4e38f;
    for (int s = 0; s < kS; ++s) m = fmaxf(m, se[s]);
    float sum = 0.0f;
    for (int s = 0; s < kS; ++s) { float e = __expf(se[s] - m); se[s] = e; sum += e; }
    float inv = 1.0f / sum;
    for (int s = 0; s < kS; ++s) se[s] *= inv;
  }
  __syncthreads();
  for (int s = tid; s < kS; s += 128) {
    float a = se[s];
    o_cfull[((size_t)t * kB + b) * kS + s] = a;
    o_clast[b * kS + s] = a;
  }
  for (int e = tid; e < kE; e += 128) {
    float acc = 0.0f;
    for (int s = 0; s < kS; ++s)
      acc += se[s] * context[((size_t)s * kB + b) * kE + e];
    ctxf[b * kE + e] = acc;
    ctxbf[b * kE + e] = (__bf16)acc;
    o_ctxf[b * kE + e] = acc;
  }
  __syncthreads();
  float p = 0.0f;
  for (int j = tid; j < kH + kE; j += 128) {
    float xv = (j < kH) ? h1f[b * kH + j] : ctxf[b * kE + (j - kH)];
    p += Wcopy[j] * xv;
  }
  sr[tid] = p;
  __syncthreads();
  for (int off = 64; off > 0; off >>= 1) {
    if (tid < off) sr[tid] += sr[tid + off];
    __syncthreads();
  }
  if (tid == 0) o_copy[(size_t)t * kB + b] = sigm(sr[0] + bcopy[0]);
}

__global__ void k_maxout(int t, const float* __restrict__ readf, float* __restrict__ gout) {
  int i = blockIdx.x * 256 + threadIdx.x;      // < 64*256
  int b = i >> 8, j = i & 255;
  float a = readf[b * kH + 2 * j];
  float c = readf[b * kH + 2 * j + 1];
  gout[((size_t)t * kB + b) * (kH / 2) + j] = fmaxf(a, c);
}

// ---------------- host ----------------
extern "C" void kernel_launch(void* const* d_in, const int* in_sizes, int n_in,
                              void* d_out, int out_size, void* d_ws, size_t ws_size,
                              hipStream_t stream) {
  const int*   input    = (const int*)  d_in[0];
  const float* hidden   = (const float*)d_in[1];
  const float* context  = (const float*)d_in[2];
  const float* mask     = (const float*)d_in[3];
  const float* init_att = (const float*)d_in[4];
  const float* emb_tab  = (const float*)d_in[5];
  const float* W_ih0 = (const float*)d_in[6];
  const float* W_hh0 = (const float*)d_in[7];
  const float* b_ih0 = (const float*)d_in[8];
  const float* b_hh0 = (const float*)d_in[9];
  const float* W_ih1 = (const float*)d_in[10];
  const float* W_hh1 = (const float*)d_in[11];
  const float* b_ih1 = (const float*)d_in[12];
  const float* b_hh1 = (const float*)d_in[13];
  const float* W_pre = (const float*)d_in[14];
  const float* b_pre = (const float*)d_in[15];
  const float* W_q   = (const float*)d_in[16];
  const float* v_att = (const float*)d_in[17];
  const float* Wcopy = (const float*)d_in[18];
  const float* bcopy = (const float*)d_in[19];
  const float* W_read= (const float*)d_in[20];
  const float* b_read= (const float*)d_in[21];

  // ---- workspace carve-up ----
  char* ws = (char*)d_ws;
  size_t off = 0;
  auto take = [&](size_t bytes) -> char* {
    char* p = ws + off;
    off = (off + bytes + 255) & ~(size_t)255;
    return p;
  };
  __bf16* wih0b  = (__bf16*)take((size_t)kH3 * (kDw + kE) * 2);
  __bf16* whh0b  = (__bf16*)take((size_t)kH3 * kH * 2);
  __bf16* wih1b  = (__bf16*)take((size_t)kH3 * kH * 2);
  __bf16* whh1b  = (__bf16*)take((size_t)kH3 * kH * 2);
  __bf16* wqb    = (__bf16*)take((size_t)kA * kH * 2);
  __bf16* wreadb = (__bf16*)take((size_t)kH * (kE + kH + kDw) * 2);
  __bf16* wpreb  = (__bf16*)take((size_t)kA * kE * 2);
  __bf16* embbf  = (__bf16*)take((size_t)kT * kB * kDw * 2);
  __bf16* ctxbf  = (__bf16*)take((size_t)kB * kS * kE * 2);
  float*  pre    = (float*) take((size_t)kB * kS * kA * 4);
  float*  h0f[2] = { (float*)take(kB * kH * 4), (float*)take(kB * kH * 4) };
  float*  h1f[2] = { (float*)take(kB * kH * 4), (float*)take(kB * kH * 4) };
  __bf16* h0b[2] = { (__bf16*)take(kB * kH * 2), (__bf16*)take(kB * kH * 2) };
  __bf16* h1b[2] = { (__bf16*)take(kB * kH * 2), (__bf16*)take(kB * kH * 2) };
  float*  ctxf   = (float*) take(kB * kE * 4);
  __bf16* ctxbs  = (__bf16*)take(kB * kE * 2);
  float*  qf     = (float*) take(kB * kA * 4);
  float*  readf  = (float*) take(kB * kH * 4);

  // ---- output carve-up (reference return order, flattened) ----
  float* out = (float*)d_out;
  float* o_gout  = out;                                   // (T,B,256)
  float* o_cout  = o_gout + (size_t)kT * kB * (kH / 2);   // (T,B,S)
  float* o_copy  = o_cout + (size_t)kT * kB * kS;         // (T,B,1)
  float* o_hidf  = o_copy + (size_t)kT * kB;              // (2,B,H)
  float* o_clast = o_hidf + (size_t)2 * kB * kH;          // (B,S)
  float* o_ctxf  = o_clast + (size_t)kB * kS;             // (B,E)
  float* o_ghid  = o_ctxf + (size_t)kB * kE;              // (T,2,B,H)

  auto grd = [](size_t n) { return dim3((unsigned)((n + 255) / 256)); };

  // ---- prologue: bf16 conversions (weights keep natural (N,K) layout) ----
  k_conv_bf<<<grd((size_t)kH3 * 1024), 256, 0, stream>>>(W_ih0, wih0b, (size_t)kH3 * 1024);
  k_conv_bf<<<grd((size_t)kH3 * kH),   256, 0, stream>>>(W_hh0, whh0b, (size_t)kH3 * kH);
  k_conv_bf<<<grd((size_t)kH3 * kH),   256, 0, stream>>>(W_ih1, wih1b, (size_t)kH3 * kH);
  k_conv_bf<<<grd((size_t)kH3 * kH),   256, 0, stream>>>(W_hh1, whh1b, (size_t)kH3 * kH);
  k_conv_bf<<<grd((size_t)kA * kH),    256, 0, stream>>>(W_q,   wqb,   (size_t)kA * kH);
  k_conv_bf<<<grd((size_t)kH * 1536),  256, 0, stream>>>(W_read, wreadb, (size_t)kH * 1536);
  k_conv_bf<<<grd((size_t)kA * kE),    256, 0, stream>>>(W_pre, wpreb, (size_t)kA * kE);
  k_embed<<<grd((size_t)kT * kB * kDw), 256, 0, stream>>>(input, emb_tab, embbf);
  k_ctx_bf<<<grd((size_t)kB * kS * kE), 256, 0, stream>>>(context, ctxbf);
  k_init<<<grd(kB * kH), 256, 0, stream>>>(hidden, init_att, h0f[0], h1f[0],
                                           h0b[0], h1b[0], ctxf, ctxbs);

  // pre = ctx_bse @ W_pre^T + b_pre   (M = B*S = 6400, N = A = 512, K = E)
  {
    int waves = (kB * kS / 16) * (kA / 64);   // 3200
    k_gemm<<<dim3(waves / 8), 256, 0, stream>>>(ctxbf, ctxbf, ctxbf, 1 << 30, 1 << 30,
                                                kE, wpreb, kA, b_pre, pre);
  }

  // ---- sequential decode ----
  for (int t = 0; t < kT; ++t) {
    int pi = t & 1, ni = (t + 1) & 1;
    // layer 0: x = [emb_t | ctx]
    k_gru<<<dim3(4), 256, 0, stream>>>(embbf + (size_t)t * kB * kDw, ctxbs, kDw,
                                       kDw + kE, wih0b, whh0b, b_ih0, b_hh0,
                                       h0f[pi], h0b[pi], h0f[ni], h0b[ni],
                                       o_ghid + ((size_t)t * 2 + 0) * kB * kH,
                                       o_hidf + 0);
    // layer 1: input = h0
    k_gru<<<dim3(4), 256, 0, stream>>>(h0b[ni], h0b[ni], 1 << 30,
                                       kH, wih1b, whh1b, b_ih1, b_hh1,
                                       h1f[pi], h1b[pi], h1f[ni], h1b[ni],
                                       o_ghid + ((size_t)t * 2 + 1) * kB * kH,
                                       o_hidf + (size_t)kB * kH);
    // q = h1 @ W_q^T
    k_gemm<<<dim3(4), 256, 0, stream>>>(h1b[ni], h1b[ni], h1b[ni], 1 << 30, 1 << 30,
                                        kH, wqb, kA, nullptr, qf);
    // attention + softmax + context + copy gate
    k_attn<<<dim3(kB), 128, 0, stream>>>(t, pre, qf, v_att, mask, context,
                                         h1f[ni], Wcopy, bcopy, ctxf, ctxbs,
                                         o_cout, o_clast, o_ctxf, o_copy);
    // readout = [emb | h1 | ctx] @ W_read^T + b_read
    k_gemm<<<dim3(4), 256, 0, stream>>>(embbf + (size_t)t * kB * kDw, h1b[ni], ctxbs,
                                        kDw, kDw + kH, kDw + kH + kE,
                                        wreadb, kH, b_read, readf);
    // maxout pool(2) -> g_out[t]
    k_maxout<<<dim3(kB), 256, 0, stream>>>(t, readf, o_gout);
  }
}
